// AttentiveTransformer_26792005992684
// MI455X (gfx1250) — matrix operand order repaired and checked by reading, not compiled, not run on hardware
//
#include <hip/hip_runtime.h>
#include <hip/hip_bf16.h>
#include <stdint.h>

// ---------------------------------------------------------------------------
// Problem constants (from reference)
// ---------------------------------------------------------------------------
#define B_ROWS 16384
#define K_DIM  2048
#define N_DIM  2048
#define VBS    128
#define EPSBN  1e-5f

// GEMM tiling
#define BM 128
#define BN 128
#define BK 32
#define LDSKU (BK / 2 + 4)   // uint stride per LDS row: 16 data uints + 4 pad = 20
#define TILEU (BM * LDSKU)   // uints per plane per buffer

typedef __attribute__((ext_vector_type(16))) __bf16 v16bf;
typedef __attribute__((ext_vector_type(8)))  float  v8f;
typedef int v4i __attribute__((vector_size(16)));

union Frag16 { uint32_t u[8]; v16bf v; };
union Acc8   { float    f[8]; v8f   v; };

// ---------------------------------------------------------------------------
// CDNA5 async global->LDS copy (ASYNCcnt-tracked), with sync fallback
// ---------------------------------------------------------------------------
#if defined(__has_builtin)
# if __has_builtin(__builtin_amdgcn_global_load_async_to_lds_b128)
#  define HAVE_ASYNC_COPY 1
# endif
# if __has_builtin(__builtin_amdgcn_s_wait_asynccnt)
#  define HAVE_WAIT_ASYNC_BUILTIN 1
# endif
#endif

#if defined(HAVE_ASYNC_COPY)
__device__ __forceinline__ void async_cp16(const uint32_t* g, uint32_t* l) {
  // (global src, lds dst, imm offset, imm cpol); generic v4i* per clang proto
  __builtin_amdgcn_global_load_async_to_lds_b128((v4i*)g, (v4i*)l, 0, 0);
}
__device__ __forceinline__ void wait_async0() {
#if defined(HAVE_WAIT_ASYNC_BUILTIN)
  __builtin_amdgcn_s_wait_asynccnt(0);
#else
  asm volatile("s_wait_asynccnt 0x0" ::: "memory");
#endif
}
#endif

// Issue (or perform) the cooperative fill of one K-step tile into one buffer.
__device__ __forceinline__ void tile_fill(const uint32_t* __restrict__ xh,
                                          const uint32_t* __restrict__ xl,
                                          const uint32_t* __restrict__ wh,
                                          const uint32_t* __restrict__ wl,
                                          uint32_t* sAh, uint32_t* sAl,
                                          uint32_t* sBh, uint32_t* sBl,
                                          int tid, int blockM, int blockN,
                                          int k0u, int KU) {
#pragma unroll
  for (int j = 0; j < 2; ++j) {
    int c    = tid + j * 256;          // 0..511 chunks per plane
    int row  = c >> 2;                 // 0..127
    int colu = (c & 3) * 4;            // 0,4,8,12
    size_t ga = (size_t)(blockM + row) * KU + k0u + colu;
    size_t gb = (size_t)(blockN + row) * KU + k0u + colu;
    int so = row * LDSKU + colu;
#if defined(HAVE_ASYNC_COPY)
    async_cp16(&xh[ga], &sAh[so]);
    async_cp16(&xl[ga], &sAl[so]);
    async_cp16(&wh[gb], &sBh[so]);
    async_cp16(&wl[gb], &sBl[so]);
#else
    *(uint4*)&sAh[so] = *(const uint4*)&xh[ga];
    *(uint4*)&sAl[so] = *(const uint4*)&xl[ga];
    *(uint4*)&sBh[so] = *(const uint4*)&wh[gb];
    *(uint4*)&sBl[so] = *(const uint4*)&wl[gb];
#endif
  }
}

// ---------------------------------------------------------------------------
// fp32 -> bf16 hi/lo split
// ---------------------------------------------------------------------------
__device__ inline uint32_t f2bf_round(float f) {
  uint32_t u = __float_as_uint(f);
  return (u + 0x7FFFu + ((u >> 16) & 1u)) >> 16;
}

__global__ void split_fp32_to_bf16(const float* __restrict__ src,
                                   uint32_t* __restrict__ hi,
                                   uint32_t* __restrict__ lo, int n2) {
  int i = blockIdx.x * blockDim.x + threadIdx.x;
  if (i >= n2) return;
  float2 f = ((const float2*)src)[i];

  uint32_t h0 = f2bf_round(f.x);
  float    r0 = f.x - __uint_as_float(h0 << 16);
  uint32_t l0 = f2bf_round(r0);
  uint32_t h1 = f2bf_round(f.y);
  float    r1 = f.y - __uint_as_float(h1 << 16);
  uint32_t l1 = f2bf_round(r1);

  hi[i] = (h0 & 0xFFFFu) | (h1 << 16);
  lo[i] = (l0 & 0xFFFFu) | (l1 << 16);
}

// ---------------------------------------------------------------------------
// h = x @ W^T via split-bf16 WMMA (hi*hi + hi*lo + lo*hi), fp32 accumulate.
// Block = 256 threads (8 waves, 2Mx4N), block tile 128x128, wave tile 64x32.
// Double-buffered LDS, async global->LDS prefetch.
// ---------------------------------------------------------------------------
__launch_bounds__(256)
__global__ void gemm_split_bf16(const uint32_t* __restrict__ xh,
                                const uint32_t* __restrict__ xl,
                                const uint32_t* __restrict__ wh,
                                const uint32_t* __restrict__ wl,
                                float* __restrict__ hout) {
  __shared__ uint32_t sAh[2 * TILEU];
  __shared__ uint32_t sAl[2 * TILEU];
  __shared__ uint32_t sBh[2 * TILEU];
  __shared__ uint32_t sBl[2 * TILEU];

  const int tid  = threadIdx.x;
  const int lane = tid & 31;
  const int wid  = tid >> 5;       // 0..7
  const int wr   = wid & 1;        // wave row (M): 64 rows each
  const int wc   = wid >> 1;       // wave col (N): 32 cols each
  const int half = lane >> 4;
  const int l16  = lane & 15;

  const int blockM = blockIdx.y * BM;
  const int blockN = blockIdx.x * BN;
  const int KU = K_DIM / 2;        // uints per global row
  const int NK = K_DIM / BK;       // 64 K-steps

  Acc8 acc[4][2];
#pragma unroll
  for (int mi = 0; mi < 4; ++mi)
#pragma unroll
    for (int ni = 0; ni < 2; ++ni)
#pragma unroll
      for (int r = 0; r < 8; ++r) acc[mi][ni].f[r] = 0.0f;

  // Prologue: fill buffer 0 with K-step 0.
  tile_fill(xh, xl, wh, wl, sAh, sAl, sBh, sBl, tid, blockM, blockN, 0, KU);

#pragma unroll 1
  for (int ki = 0; ki < NK; ++ki) {
    const int buf  = ki & 1;
    const int bofs = buf * TILEU;
#if defined(HAVE_ASYNC_COPY)
    wait_async0();                 // data for this buffer has landed in LDS
#endif
    __syncthreads();               // ...for every wave; prev compute done

    if (ki + 1 < NK) {
      // Prefetch next K-step into the other buffer. Safe: all waves finished
      // reading it (iteration ki-1) before the barrier above.
      const int nofs = (buf ^ 1) * TILEU;
      tile_fill(xh, xl, wh, wl, sAh + nofs, sAl + nofs, sBh + nofs, sBl + nofs,
                tid, blockM, blockN, (ki + 1) * (BK / 2), KU);
    }

    // A fragments (hi/lo) for this wave's 4 M-subtiles.
    // ISA 16-bit A 16x32 layout: lanes0-15 K in {0-7,16-23}, lanes16-31
    // K in {8-15,24-31} -> two contiguous 16B chunks per lane.
    Frag16 Ah[4], Al[4];
#pragma unroll
    for (int mi = 0; mi < 4; ++mi) {
      int m = wr * 64 + mi * 16 + l16;
      const uint32_t* rh = &sAh[bofs + m * LDSKU];
      const uint32_t* rl = &sAl[bofs + m * LDSKU];
      uint4 h0 = *(const uint4*)&rh[half * 4];
      uint4 h1 = *(const uint4*)&rh[8 + half * 4];
      uint4 l0 = *(const uint4*)&rl[half * 4];
      uint4 l1 = *(const uint4*)&rl[8 + half * 4];
      Ah[mi].u[0] = h0.x; Ah[mi].u[1] = h0.y; Ah[mi].u[2] = h0.z; Ah[mi].u[3] = h0.w;
      Ah[mi].u[4] = h1.x; Ah[mi].u[5] = h1.y; Ah[mi].u[6] = h1.z; Ah[mi].u[7] = h1.w;
      Al[mi].u[0] = l0.x; Al[mi].u[1] = l0.y; Al[mi].u[2] = l0.z; Al[mi].u[3] = l0.w;
      Al[mi].u[4] = l1.x; Al[mi].u[5] = l1.y; Al[mi].u[6] = l1.z; Al[mi].u[7] = l1.w;
    }

#pragma unroll
    for (int ni = 0; ni < 2; ++ni) {
      // B 32x16 layout (matches A): lanes0-15 K in {0-7,16-23}, lanes16-31
      // K in {8-15,24-31} -> two contiguous 16B chunks per lane.
      int n = wc * 32 + ni * 16 + l16;
      Frag16 Bh, Bl;
      const uint32_t* rh = &sBh[bofs + n * LDSKU];
      const uint32_t* rl = &sBl[bofs + n * LDSKU];
      uint4 b0 = *(const uint4*)&rh[half * 4];
      uint4 b1 = *(const uint4*)&rh[8 + half * 4];
      uint4 c0 = *(const uint4*)&rl[half * 4];
      uint4 c1 = *(const uint4*)&rl[8 + half * 4];
      Bh.u[0] = b0.x; Bh.u[1] = b0.y; Bh.u[2] = b0.z; Bh.u[3] = b0.w;
      Bh.u[4] = b1.x; Bh.u[5] = b1.y; Bh.u[6] = b1.z; Bh.u[7] = b1.w;
      Bl.u[0] = c0.x; Bl.u[1] = c0.y; Bl.u[2] = c0.z; Bl.u[3] = c0.w;
      Bl.u[4] = c1.x; Bl.u[5] = c1.y; Bl.u[6] = c1.z; Bl.u[7] = c1.w;

#pragma unroll
      for (int mi = 0; mi < 4; ++mi) {
        acc[mi][ni].v = __builtin_amdgcn_wmma_f32_16x16x32_bf16(
            false, Ah[mi].v, false, Bh.v, (short)0, acc[mi][ni].v, false, false);
        acc[mi][ni].v = __builtin_amdgcn_wmma_f32_16x16x32_bf16(
            false, Ah[mi].v, false, Bl.v, (short)0, acc[mi][ni].v, false, false);
        acc[mi][ni].v = __builtin_amdgcn_wmma_f32_16x16x32_bf16(
            false, Al[mi].v, false, Bh.v, (short)0, acc[mi][ni].v, false, false);
      }
    }
  }

  // Epilogue: C layout 16x16 f32 -> VGPR r holds M=r (lanes0-15) / M=r+8.
#pragma unroll
  for (int mi = 0; mi < 4; ++mi)
#pragma unroll
    for (int ni = 0; ni < 2; ++ni)
#pragma unroll
      for (int r = 0; r < 8; ++r) {
        int m = blockM + wr * 64 + mi * 16 + half * 8 + r;
        int n = blockN + wc * 32 + ni * 16 + l16;
        hout[(size_t)m * N_DIM + n] = acc[mi][ni].f[r];
      }
}

// ---------------------------------------------------------------------------
// Ghost-BN stats: per (virtual batch, column) -> fused scale/shift
// ---------------------------------------------------------------------------
__global__ void bn_stats(const float* __restrict__ h,
                         const float* __restrict__ gamma,
                         const float* __restrict__ beta,
                         float* __restrict__ scale,
                         float* __restrict__ shift) {
  int col = blockIdx.x * blockDim.x + threadIdx.x;
  int vb  = blockIdx.y;
  const float* p = h + (size_t)vb * VBS * N_DIM + col;
  float s = 0.f, ss = 0.f;
#pragma unroll 4
  for (int r = 0; r < VBS; ++r) {
    float v = p[(size_t)r * N_DIM];
    s += v;
    ss += v * v;
  }
  float mean = s * (1.0f / VBS);
  float var  = ss * (1.0f / VBS) - mean * mean;
  float sc   = gamma[col] * __frsqrt_rn(var + EPSBN);
  scale[vb * N_DIM + col] = sc;
  shift[vb * N_DIM + col] = beta[col] - mean * sc;
}

// ---------------------------------------------------------------------------
// Row-wise sparsemax via Newton on f(tau) = sum(relu(z - tau)) - 1.
// Convex, decreasing, piecewise linear: Newton from tau0 = max(z) - 1
// approaches the root monotonically from the left; exact on convergence.
// ---------------------------------------------------------------------------
__device__ inline float waveSum(float v) {
#pragma unroll
  for (int o = 16; o > 0; o >>= 1) v += __shfl_xor(v, o, 32);
  return v;
}
__device__ inline float waveMax(float v) {
#pragma unroll
  for (int o = 16; o > 0; o >>= 1) v = fmaxf(v, __shfl_xor(v, o, 32));
  return v;
}

__launch_bounds__(256)
__global__ void sparsemax_rows(float* __restrict__ h,            // in/out (d_out)
                               const float* __restrict__ prior,
                               const float* __restrict__ scale,
                               const float* __restrict__ shift) {
  __shared__ float redA[8];
  __shared__ float redB[8];
  const int row  = blockIdx.x;
  const int vb   = row >> 7;          // / VBS
  const int tid  = threadIdx.x;
  const int lane = tid & 31;
  const int wid  = tid >> 5;

  float* hp = h + (size_t)row * N_DIM;
  const float* pp  = prior + (size_t)row * N_DIM;
  const float* scp = scale + (size_t)vb * N_DIM;
  const float* shp = shift + (size_t)vb * N_DIM;

  float z[8];
  float zmax = -3.0e38f;
#pragma unroll
  for (int j = 0; j < 8; ++j) {
    int c = tid + j * 256;
    float v = fmaf(hp[c], scp[c], shp[c]) * pp[c];
    z[j] = v;
    zmax = fmaxf(zmax, v);
  }

  zmax = waveMax(zmax);
  if (lane == 0) redA[wid] = zmax;
  __syncthreads();
  float m = redA[0];
#pragma unroll
  for (int w = 1; w < 8; ++w) m = fmaxf(m, redA[w]);
  __syncthreads();

  float tau = m - 1.0f;
#pragma unroll 1
  for (int it = 0; it < 32; ++it) {
    float s = 0.f, cnt = 0.f;
#pragma unroll
    for (int j = 0; j < 8; ++j) {
      float d = z[j] - tau;
      if (d > 0.f) { s += d; cnt += 1.f; }
    }
    s = waveSum(s);
    cnt = waveSum(cnt);
    if (lane == 0) { redA[wid] = s; redB[wid] = cnt; }
    __syncthreads();
    float S = 0.f, C = 0.f;
#pragma unroll
    for (int w = 0; w < 8; ++w) { S += redA[w]; C += redB[w]; }
    __syncthreads();
    if (C < 1.0f) C = 1.0f;
    tau += (S - 1.0f) / C;
  }

#pragma unroll
  for (int j = 0; j < 8; ++j) {
    int c = tid + j * 256;
    hp[c] = fmaxf(z[j] - tau, 0.0f);
  }
}

// ---------------------------------------------------------------------------
// Launch. Workspace (~153 MB):
//   xh (67MB) | xl (67MB) | wh (8.4MB) | wl (8.4MB) | scale (1MB) | shift (1MB)
// d_out doubles as the fp32 h buffer between GEMM and sparsemax.
// ---------------------------------------------------------------------------
extern "C" void kernel_launch(void* const* d_in, const int* in_sizes, int n_in,
                              void* d_out, int out_size, void* d_ws, size_t ws_size,
                              hipStream_t stream) {
  const float* prior = (const float*)d_in[0];
  const float* x     = (const float*)d_in[1];
  const float* W     = (const float*)d_in[2];
  const float* gamma = (const float*)d_in[3];
  const float* beta  = (const float*)d_in[4];
  float* out = (float*)d_out;

  const size_t nx = (size_t)B_ROWS * K_DIM;
  const size_t nw = (size_t)N_DIM * K_DIM;
  const int    nvb = B_ROWS / VBS;

  char* ws = (char*)d_ws;
  uint32_t* xh = (uint32_t*)ws;  ws += nx * 2;
  uint32_t* xl = (uint32_t*)ws;  ws += nx * 2;
  uint32_t* wh = (uint32_t*)ws;  ws += nw * 2;
  uint32_t* wl = (uint32_t*)ws;  ws += nw * 2;
  float* scale = (float*)ws;     ws += (size_t)nvb * N_DIM * sizeof(float);
  float* shift = (float*)ws;

  split_fp32_to_bf16<<<(int)(nx / 2 / 256), 256, 0, stream>>>(x, xh, xl, (int)(nx / 2));
  split_fp32_to_bf16<<<(int)(nw / 2 / 256), 256, 0, stream>>>(W, wh, wl, (int)(nw / 2));

  gemm_split_bf16<<<dim3(N_DIM / BN, B_ROWS / BM), 256, 0, stream>>>(xh, xl, wh, wl, out);

  bn_stats<<<dim3(N_DIM / 256, nvb), 256, 0, stream>>>(out, gamma, beta, scale, shift);

  sparsemax_rows<<<B_ROWS, 256, 0, stream>>>(out, prior, scale, shift);
}
